// GNNGaussianencoder_49881750175987
// MI455X (gfx1250) — matrix operand order repaired and checked by reading, not compiled, not run on hardware
//
#include <hip/hip_runtime.h>
#include <hip/hip_bf16.h>
#include <math.h>

// ---------------- problem constants (from reference) ----------------
#define NN   100000
#define EE   1600000
#define BB   2000
#define NWc  256
#define ENCc 512
#define KT   100      // num_topic
#define KPAD 128      // 8 x 16 tiles -> 4 tile-pairs for WMMA
#define EPSc 1e-5f
#define LPAD 8        // LDS row padding (bf16 elems) to break bank aliasing

typedef __bf16 bh;
typedef __attribute__((ext_vector_type(16))) __bf16 v16bf;
typedef __attribute__((ext_vector_type(8)))  __bf16 v8bf;
typedef __attribute__((ext_vector_type(8)))  float  v8f;

__device__ __forceinline__ v8f wmma_bf16(v16bf a, v16bf b, v8f c) {
  return __builtin_amdgcn_wmma_f32_16x16x32_bf16(false, a, false, b, (short)0, c,
                                                 false, false);
}

// ---------------- elementwise / prep kernels ----------------

// x = word_vec[idx_x]; also bf16 copy and aggr = idx_w * x (self-loop folded in)
__global__ void k_embed(const int* __restrict__ idx, const float* __restrict__ wv,
                        const float* __restrict__ w0p,
                        float* __restrict__ xf, bh* __restrict__ xb,
                        float* __restrict__ aggr) {
  long t = (long)blockIdx.x * 256 + threadIdx.x;       // N*64 threads, float4 each
  if (t >= (long)NN * 64) return;
  int n = (int)(t >> 6), c4 = ((int)t & 63) << 2;
  const float4 v = *(const float4*)(wv + (long)idx[n] * NWc + c4);
  float w0 = w0p[0];
  *(float4*)(xf + (long)n * NWc + c4) = v;
  float4 av; av.x = w0 * v.x; av.y = w0 * v.y; av.z = w0 * v.z; av.w = w0 * v.w;
  *(float4*)(aggr + (long)n * NWc + c4) = av;
  bh* xp = xb + (long)n * NWc + c4;
  xp[0] = (bh)v.x; xp[1] = (bh)v.y; xp[2] = (bh)v.z; xp[3] = (bh)v.w;
}

__global__ void k_zero(float* __restrict__ p, long n) {
  long t = (long)blockIdx.x * 256 + threadIdx.x;
  if (t < n) p[t] = 0.f;
}

// scatter: aggr[dst] += w * x[src]   (64 threads/edge, float4 per thread)
__global__ void k_edges(const int* __restrict__ src, const int* __restrict__ dst,
                        const float* __restrict__ ew, const float* __restrict__ xf,
                        float* __restrict__ aggr) {
  long t = (long)blockIdx.x * 256 + threadIdx.x;
  long e = t >> 6;
  if (e >= (long)EE) return;
  int c4 = ((int)t & 63) << 2;
  int s = src[e], d = dst[e];
  float w = ew[e];
  const float4 v = *(const float4*)(xf + (long)s * NWc + c4);
  float* ap = aggr + (long)d * NWc + c4;
  atomicAdd(ap + 0, w * v.x);
  atomicAdd(ap + 1, w * v.y);
  atomicAdd(ap + 2, w * v.z);
  atomicAdd(ap + 3, w * v.w);
}

__global__ void k_cvt(const float* __restrict__ s, bh* __restrict__ d, long n) {
  long t = (long)blockIdx.x * 256 + threadIdx.x;
  if (t < n) d[t] = (bh)s[t];
}

// Wcat[o, 0:256]=Wrel[o], Wcat[o, 256:512]=Wroot[o]  (bf16)
__global__ void k_cat2(const float* __restrict__ A, const float* __restrict__ Bm,
                       bh* __restrict__ d, int rows, int half) {
  long t = (long)blockIdx.x * 256 + threadIdx.x;
  long tot = (long)rows * 2 * half;
  if (t >= tot) return;
  int o = (int)(t / (2 * half)), c = (int)(t % (2 * half));
  float v = (c < half) ? A[(long)o * half + c] : Bm[(long)o * half + (c - half)];
  d[t] = (bh)v;
}

// Wphi [100,1024] -> bf16 padded to [128,1024] with zero rows
__global__ void k_padphi(const float* __restrict__ W, bh* __restrict__ d) {
  long t = (long)blockIdx.x * 256 + threadIdx.x;
  if (t >= (long)KPAD * 1024) return;
  int o = (int)(t >> 10), c = (int)t & 1023;
  d[t] = (bh)((o < KT) ? W[(long)o * 1024 + c] : 0.f);
}

// ---------------- LDS-staged split-A WMMA GEMM ----------------
// C = [A1 | A2(rowmap)] * W^T + bias. Block stages its 16-row A stripe (concat
// resolved + rowmap gathered) into LDS once; each wave computes a 16x32 slab
// (two 16x16 tiles, dual accumulators -> independent back-to-back WMMAs).
__global__ __launch_bounds__(256) void k_gemm(
    const bh* __restrict__ A1, int lda1,
    const bh* __restrict__ A2, int lda2, const int* __restrict__ rowmap,
    int ksplit, int K,
    const bh* __restrict__ W, const float* __restrict__ bias,
    float* __restrict__ C, int ldc, int npairs, int nreal) {
  __shared__ __align__(16) bh As[16 * (1024 + LPAD)];
  const int stride = K + LPAD;
  const int m0 = blockIdx.x << 4;

  // cooperative stage: 16 rows x K bf16, 8-elem (16B) chunks
  const int CK = K >> 3;
  for (int q = threadIdx.x; q < 16 * CK; q += 256) {
    int r = q / CK;
    int kc = (q - r * CK) << 3;
    int grow = m0 + r;
    const bh* srcp;
    if (kc < ksplit) {
      srcp = A1 + (long)grow * lda1 + kc;
    } else {
      int g2 = rowmap ? rowmap[grow] : grow;
      srcp = A2 + (long)g2 * lda2 + (kc - ksplit);
    }
    *(v8bf*)(&As[r * stride + kc]) = *(const v8bf*)srcp;
  }
  __syncthreads();

  int wave = threadIdx.x >> 5;
  int lane = threadIdx.x & 31;
  int pair = blockIdx.y * 8 + wave;
  if (pair < npairs) {                          // wave-uniform guard
    int lane15 = lane & 15;
    int kAoff = (lane >> 4) << 4;               // 0 / 16 (contiguous k = 16h + s for A, same as B)
    int kBoff = (lane >> 4) << 4;               // 0 / 16 (contiguous k = 16h + s for B, same as A)
    int col0 = (pair << 5) + lane15;
    int col1 = col0 + 16;
    const bh* as = As + lane15 * stride + kAoff;
    const bh* w0p = W + (long)col0 * K + kBoff;
    const bh* w1p = W + (long)col1 * K + kBoff;
    v8f acc0 = {0.f, 0.f, 0.f, 0.f, 0.f, 0.f, 0.f, 0.f};
    v8f acc1 = {0.f, 0.f, 0.f, 0.f, 0.f, 0.f, 0.f, 0.f};
    for (int k0 = 0; k0 < K; k0 += 32) {
      union { v16bf v; v8bf h[2]; } af;
      af.h[0] = *(const v8bf*)(as + k0);        // ds_load_b128
      af.h[1] = *(const v8bf*)(as + k0 + 8);
      v16bf b0 = *(const v16bf*)(w0p + k0);     // global, L2-resident weights
      v16bf b1 = *(const v16bf*)(w1p + k0);
      acc0 = wmma_bf16(af.v, b0, acc0);
      acc1 = wmma_bf16(af.v, b1, acc1);
    }
    int rbase = m0 + ((lane >> 4) << 3);
    if (col0 < nreal) {
      float bb = bias ? bias[col0] : 0.f;
      #pragma unroll
      for (int r = 0; r < 8; ++r)
        C[(long)(rbase + r) * ldc + col0] = acc0[r] + bb;
    }
    if (col1 < nreal) {
      float bb = bias ? bias[col1] : 0.f;
      #pragma unroll
      for (int r = 0; r < 8; ++r)
        C[(long)(rbase + r) * ldc + col1] = acc1[r] + bb;
    }
  }
}

// ---------------- fused GLU + pooled segment-sum ----------------
// enc2 = sigmoid(A*W1^T+b1) * tanh(A*W2^T+b2); pooled[x_batch[row]] += enc2.
// LDS-staged A; each wave: 2 cols x 2 gates = 4 independent WMMAs per K-step.
__global__ __launch_bounds__(256) void k_glu_pool(
    const bh* __restrict__ A,                 // [N, 512] bf16 (enc1 concat)
    const bh* __restrict__ W1b, const bh* __restrict__ W2b,  // [512,512] bf16
    const float* __restrict__ b1, const float* __restrict__ b2,
    const int* __restrict__ xb, float* __restrict__ pooled) {
  __shared__ __align__(16) bh As[16 * (ENCc + LPAD)];
  const int stride = ENCc + LPAD;
  const int m0 = blockIdx.x << 4;

  const int CK = ENCc >> 3;
  for (int q = threadIdx.x; q < 16 * CK; q += 256) {
    int r = q / CK;
    int kc = (q - r * CK) << 3;
    *(v8bf*)(&As[r * stride + kc]) = *(const v8bf*)(A + (long)(m0 + r) * ENCc + kc);
  }
  __syncthreads();

  int wave = threadIdx.x >> 5;
  int lane = threadIdx.x & 31;
  int pair = blockIdx.y * 8 + wave;           // 16 pairs cover 512 cols
  int lane15 = lane & 15;
  int kAoff = (lane >> 4) << 4;
  int kBoff = (lane >> 4) << 4;
  int col0 = (pair << 5) + lane15;
  int col1 = col0 + 16;
  const bh* as = As + lane15 * stride + kAoff;
  const bh* w10 = W1b + (long)col0 * ENCc + kBoff;
  const bh* w11 = W1b + (long)col1 * ENCc + kBoff;
  const bh* w20 = W2b + (long)col0 * ENCc + kBoff;
  const bh* w21 = W2b + (long)col1 * ENCc + kBoff;
  v8f g0 = {0.f, 0.f, 0.f, 0.f, 0.f, 0.f, 0.f, 0.f};
  v8f g1 = {0.f, 0.f, 0.f, 0.f, 0.f, 0.f, 0.f, 0.f};
  v8f h0 = {0.f, 0.f, 0.f, 0.f, 0.f, 0.f, 0.f, 0.f};
  v8f h1 = {0.f, 0.f, 0.f, 0.f, 0.f, 0.f, 0.f, 0.f};
  for (int k0 = 0; k0 < ENCc; k0 += 32) {
    union { v16bf v; v8bf h[2]; } af;
    af.h[0] = *(const v8bf*)(as + k0);
    af.h[1] = *(const v8bf*)(as + k0 + 8);
    v16bf bg0 = *(const v16bf*)(w10 + k0);
    v16bf bg1 = *(const v16bf*)(w11 + k0);
    v16bf bh0 = *(const v16bf*)(w20 + k0);
    v16bf bh1 = *(const v16bf*)(w21 + k0);
    g0 = wmma_bf16(af.v, bg0, g0);
    g1 = wmma_bf16(af.v, bg1, g1);
    h0 = wmma_bf16(af.v, bh0, h0);
    h1 = wmma_bf16(af.v, bh1, h1);
  }
  float bb10 = b1[col0], bb11 = b1[col1];
  float bb20 = b2[col0], bb21 = b2[col1];
  int rbase = m0 + ((lane >> 4) << 3);
  #pragma unroll
  for (int r = 0; r < 8; ++r) {
    int doc = xb[rbase + r];
    float e0 = (1.f / (1.f + __expf(-(g0[r] + bb10)))) * tanhf(h0[r] + bb20);
    float e1 = (1.f / (1.f + __expf(-(g1[r] + bb11)))) * tanhf(h1[r] + bb21);
    atomicAdd(pooled + (long)doc * ENCc + col0, e0);
    atomicAdd(pooled + (long)doc * ENCc + col1, e1);
  }
}

// ---------------- BatchNorm stats / finalize / enc1 pack ----------------
__global__ void k_bnstats(const float* __restrict__ h, float* __restrict__ sum,
                          float* __restrict__ sumsq) {
  int c = threadIdx.x;                               // 256 channels
  long r0 = (long)blockIdx.x * 256;
  long r1 = r0 + 256; if (r1 > (long)NN) r1 = (long)NN;
  float s1 = 0.f, s2 = 0.f;
  for (long r = r0; r < r1; ++r) { float v = h[r * NWc + c]; s1 += v; s2 += v * v; }
  atomicAdd(sum + c, s1);
  atomicAdd(sumsq + c, s2);
}

__global__ void k_bnfin(const float* __restrict__ sum, const float* __restrict__ sumsq,
                        const float* __restrict__ gamma, const float* __restrict__ beta,
                        float* __restrict__ sa, float* __restrict__ sb) {
  int c = threadIdx.x;
  float mu  = sum[c] * (1.f / (float)NN);
  float var = sumsq[c] * (1.f / (float)NN) - mu * mu;
  float a = gamma[c] * rsqrtf(var + EPSc);
  sa[c] = a;
  sb[c] = beta[c] - mu * a;
}

// enc1cat_bf[n, 0:256] = tanh(a*h+b); enc1cat_bf[n, 256:512] = x_bf
__global__ void k_enc1(const float* __restrict__ h, const bh* __restrict__ xb,
                       const float* __restrict__ sa, const float* __restrict__ sb,
                       bh* __restrict__ e1) {
  long t = (long)blockIdx.x * 256 + threadIdx.x;
  if (t >= (long)NN * ENCc) return;
  int n = (int)(t >> 9), c = (int)t & 511;
  bh v;
  if (c < NWc) v = (bh)tanhf(sa[c] * h[(long)n * NWc + c] + sb[c]);
  else         v = xb[(long)n * NWc + (c - NWc)];
  e1[t] = v;
}

// ---------------- small dense heads ----------------
__global__ void k_meanvar(const float* __restrict__ pooled,
                          const float* __restrict__ Wm, const float* __restrict__ bm,
                          const float* __restrict__ Wv, const float* __restrict__ bv,
                          float* __restrict__ out) {
  int t = blockIdx.x * 256 + threadIdx.x;
  if (t >= BB * KT) return;
  int b = t / KT, k = t % KT;
  const float* p  = pooled + (long)b * ENCc;
  const float* wm = Wm + (long)k * ENCc;
  const float* wv = Wv + (long)k * ENCc;
  float am = 0.f, av = 0.f;
  for (int j = 0; j < ENCc; ++j) { float pv = p[j]; am += pv * wm[j]; av += pv * wv[j]; }
  out[t] = am + bm[k];
  out[(long)BB * KT + t] = av + bv[k];
}

__global__ void k_softmax(const float* __restrict__ logits, float* __restrict__ out) {
  int n = blockIdx.x * 256 + threadIdx.x;
  if (n >= NN) return;
  const float* L = logits + (long)n * KT;
  float m = -1e30f;
  for (int k = 0; k < KT; ++k) m = fmaxf(m, L[k]);
  float s = 0.f;
  for (int k = 0; k < KT; ++k) s += __expf(L[k] - m);
  float inv = 1.f / s;
  float* o = out + (long)n * KT;
  for (int k = 0; k < KT; ++k) o[k] = __expf(L[k] - m) * inv;
}

// ---------------- driver ----------------
extern "C" void kernel_launch(void* const* d_in, const int* in_sizes, int n_in,
                              void* d_out, int out_size, void* d_ws, size_t ws_size,
                              hipStream_t stream) {
  const int*   idx_x   = (const int*)  d_in[0];
  const float* idx_w   = (const float*)d_in[1];
  const int*   x_batch = (const int*)  d_in[2];
  const int*   eidx    = (const int*)  d_in[3];   // [2,E]: row0=src, row1=dst
  const float* ew      = (const float*)d_in[4];
  const float* word_vec= (const float*)d_in[5];
  const float* Wrel    = (const float*)d_in[6];
  const float* brel    = (const float*)d_in[7];
  const float* Wroot   = (const float*)d_in[8];
  const float* gamma   = (const float*)d_in[9];
  const float* beta    = (const float*)d_in[10];
  const float* W1      = (const float*)d_in[11];
  const float* b1      = (const float*)d_in[12];
  const float* W2      = (const float*)d_in[13];
  const float* b2      = (const float*)d_in[14];
  const float* Wm      = (const float*)d_in[15];
  const float* bm      = (const float*)d_in[16];
  const float* Wv      = (const float*)d_in[17];
  const float* bv      = (const float*)d_in[18];
  const float* Wphi    = (const float*)d_in[19];
  const float* bphi    = (const float*)d_in[20];
  float* out = (float*)d_out;

  char* ws = (char*)d_ws;
  size_t off = 0;
  auto alloc = [&](size_t bytes) -> void* {
    void* p = ws + off;
    off += (bytes + 255) & ~(size_t)255;
    return p;
  };
  float* xf      = (float*)alloc((size_t)NN * NWc * 4);
  bh*    xb      = (bh*)   alloc((size_t)NN * NWc * 2);
  float* aggr    = (float*)alloc((size_t)NN * NWc * 4);
  bh*    aggrb   = (bh*)   alloc((size_t)NN * NWc * 2);
  float* h       = (float*)alloc((size_t)NN * NWc * 4);
  bh*    e1      = (bh*)   alloc((size_t)NN * ENCc * 2);
  float* pooled  = (float*)alloc((size_t)BB * ENCc * 4);
  bh*    pooledb = (bh*)   alloc((size_t)BB * ENCc * 2);
  float* logits  = (float*)alloc((size_t)NN * KT * 4);
  bh*    Wcat    = (bh*)   alloc((size_t)NWc * ENCc * 2);
  bh*    W1b     = (bh*)   alloc((size_t)ENCc * ENCc * 2);
  bh*    W2b     = (bh*)   alloc((size_t)ENCc * ENCc * 2);
  bh*    Wpb     = (bh*)   alloc((size_t)KPAD * 1024 * 2);
  float* sum     = (float*)alloc(NWc * 4);
  float* sumsq   = (float*)alloc(NWc * 4);
  float* sa      = (float*)alloc(NWc * 4);
  float* sb      = (float*)alloc(NWc * 4);

  const int* esrc = eidx;
  const int* edst = eidx + EE;

  // 1. embed gather + aggr = w0*x  (self-loop)
  k_embed<<<((long)NN * 64 + 255) / 256, 256, 0, stream>>>(idx_x, word_vec, idx_w, xf, xb, aggr);
  // 2. zero accumulators
  k_zero<<<((long)BB * ENCc + 255) / 256, 256, 0, stream>>>(pooled, (long)BB * ENCc);
  k_zero<<<1, 256, 0, stream>>>(sum, NWc);
  k_zero<<<1, 256, 0, stream>>>(sumsq, NWc);
  // 3. weight packs (bf16)
  k_cat2<<<((long)NWc * ENCc + 255) / 256, 256, 0, stream>>>(Wrel, Wroot, Wcat, NWc, NWc);
  k_cvt<<<((long)ENCc * ENCc + 255) / 256, 256, 0, stream>>>(W1, W1b, (long)ENCc * ENCc);
  k_cvt<<<((long)ENCc * ENCc + 255) / 256, 256, 0, stream>>>(W2, W2b, (long)ENCc * ENCc);
  k_padphi<<<((long)KPAD * 1024 + 255) / 256, 256, 0, stream>>>(Wphi, Wpb);
  // 4. edge scatter (atomic f32)
  k_edges<<<((long)EE * 64 + 255) / 256, 256, 0, stream>>>(esrc, edst, ew, xf, aggr);
  // 5. aggr -> bf16
  k_cvt<<<((long)NN * NWc + 255) / 256, 256, 0, stream>>>(aggr, aggrb, (long)NN * NWc);
  // 6. GraphConv GEMM: h = [aggr|x] @ [Wrel|Wroot]^T + brel  (K=512, 8 tile-pairs)
  k_gemm<<<dim3(NN / 16, 1), 256, 0, stream>>>(aggrb, NWc, xb, NWc, nullptr,
                                               NWc, 2 * NWc, Wcat, brel, h, NWc, 8, NWc);
  // 7. BN stats + finalize + enc1 pack
  k_bnstats<<<(NN + 255) / 256, 256, 0, stream>>>(h, sum, sumsq);
  k_bnfin<<<1, NWc, 0, stream>>>(sum, sumsq, gamma, beta, sa, sb);
  k_enc1<<<((long)NN * ENCc + 255) / 256, 256, 0, stream>>>(h, xb, sa, sb, e1);
  // 8. fused GLU + pooling (quad-WMMA per K-step, atomic segment-sum)
  k_glu_pool<<<dim3(NN / 16, 2), 256, 0, stream>>>(e1, W1b, W2b, b1, b2, x_batch, pooled);
  // 9. pooled -> bf16 for phi gather-half
  k_cvt<<<((long)BB * ENCc + 255) / 256, 256, 0, stream>>>(pooled, pooledb, (long)BB * ENCc);
  // 10. mean / logvar heads
  k_meanvar<<<(BB * KT + 255) / 256, 256, 0, stream>>>(pooled, Wm, bm, Wv, bv, out);
  // 11. phi GEMM: logits = [enc1 | pooled[x_batch]] @ Wphi^T + bphi (virtual concat, 4 pairs)
  k_gemm<<<dim3(NN / 16, 1), 256, 0, stream>>>(e1, ENCc, pooledb, ENCc, x_batch,
                                               ENCc, 1024, Wpb, bphi, logits, KT, 4, KT);
  // 12. row softmax -> phi
  k_softmax<<<(NN + 255) / 256, 256, 0, stream>>>(logits, out + 2L * BB * KT);
}
